// MultiHeadAttention_80814104642277
// MI455X (gfx1250) — matrix operand order repaired and checked by reading, not compiled, not run on hardware
//
#include <hip/hip_runtime.h>
#include <hip/hip_bf16.h>

// MI455X / gfx1250, wave32, WMMA bf16 (16x16x32, f32 accum)

typedef __attribute__((ext_vector_type(16))) __bf16 v16bf;
typedef __attribute__((ext_vector_type(8)))  __bf16 v8bf;
typedef __attribute__((ext_vector_type(8)))  float  v8f;
typedef __attribute__((ext_vector_type(4)))  int    v4i;

#define D_MODEL 1024
#define N_HEADS 16
#define D_K     64
#define BATCH   2
#define SEQ     2048

#if defined(__HIP_DEVICE_COMPILE__) && __has_builtin(__builtin_amdgcn_global_load_async_to_lds_b128)
#define HAS_ASYNC_LDS 1
#else
#define HAS_ASYNC_LDS 0
#endif

static __device__ __forceinline__ v8f wmma_bf16(v16bf a, v16bf b, v8f c) {
    return __builtin_amdgcn_wmma_f32_16x16x32_bf16(false, a, false, b, (short)0, c,
                                                   false, false);
}

static __device__ __forceinline__ void wait_async0() {
#if HAS_ASYNC_LDS
#if __has_builtin(__builtin_amdgcn_s_wait_asynccnt)
    __builtin_amdgcn_s_wait_asynccnt(0);
#else
    asm volatile("s_wait_asynccnt 0" ::: "memory");
#endif
#endif
}

#if HAS_ASYNC_LDS
// Stage one 32x64 bf16 V chunk (4096 B, contiguous in global) into LDS using
// coalesced async b128 transfers (2 per thread, 128 threads).
// Builtin signature (from diagnostics): (v4i AS1*, v4i AS3*, imm offset, imm cpol)
static __device__ __forceinline__ void stage_v(const __bf16* gsrc, __bf16* ldst, int tid)
{
    typedef __attribute__((address_space(1))) v4i gv4i;
    typedef __attribute__((address_space(3))) v4i lv4i;
    const char* g = (const char*)gsrc;
    char*       l = (char*)ldst;
    __builtin_amdgcn_global_load_async_to_lds_b128(
        (gv4i*)(g + tid * 16), (lv4i*)(l + tid * 16), 0, 0);
    __builtin_amdgcn_global_load_async_to_lds_b128(
        (gv4i*)(g + (128 + tid) * 16), (lv4i*)(l + (128 + tid) * 16), 0, 0);
}
#endif

// ---------------------------------------------------------------------------
// Kernel 1: fused Q/K/V projections.  C = X @ W^T + bias, bf16 output in
// per-head layout [b][h][s][64].  One 16x64 tile per wave (4 WMMA / K-step);
// the 64-wide N-group is exactly one head.  4096 wave-tiles per matrix.
// ---------------------------------------------------------------------------
__global__ __launch_bounds__(256)
void proj_qkv_kernel(const float* __restrict__ q, const float* __restrict__ k,
                     const float* __restrict__ v,
                     const float* __restrict__ Wq, const float* __restrict__ bq,
                     const float* __restrict__ Wk, const float* __restrict__ bk,
                     const float* __restrict__ Wv, const float* __restrict__ bv,
                     __bf16* __restrict__ qh, __bf16* __restrict__ kh,
                     __bf16* __restrict__ vh)
{
    const int lane = threadIdx.x & 31;
    const int wib  = threadIdx.x >> 5;
    const int wid  = blockIdx.x * 8 + wib;          // 0 .. 12287
    const int mat  = wid / 4096;                    // 0=q 1=k 2=v
    const int t    = wid - mat * 4096;
    const int tm   = t >> 4;                        // 0..255 (M tile)
    const int tng  = t & 15;                        // 0..15  (N group of 64 == head)

    const float* X;  const float* W;  const float* bias;  __bf16* out;
    if (mat == 0)      { X = q; W = Wq; bias = bq; out = qh; }
    else if (mat == 1) { X = k; W = Wk; bias = bk; out = kh; }
    else               { X = v; W = Wv; bias = bv; out = vh; }

    const int row0 = tm * 16;
    const int half = lane >> 4;
    const int lidx = lane & 15;

    const float* arow = X + (long)(row0 + lidx) * D_MODEL;
    // B row for sub-tile j: W[(tng*64 + j*16 + lidx)][kc + 16*half ...]
    const float* wr0  = W + (long)(tng * 64 + lidx) * D_MODEL + 16 * half;

    v8f acc[4] = {{}, {}, {}, {}};
    for (int kc = 0; kc < D_MODEL; kc += 32) {
        __builtin_prefetch(arow + kc + 64, 0, 3);
        v16bf a;
        #pragma unroll
        for (int i = 0; i < 8; ++i) {
            a[i]     = (__bf16)arow[kc +     16 * half + i];
            a[8 + i] = (__bf16)arow[kc + 8 + 16 * half + i];
        }
        #pragma unroll
        for (int j = 0; j < 4; ++j) {
            const float* wr = wr0 + (long)j * 16 * D_MODEL + kc;
            __builtin_prefetch(wr + 32, 0, 3);
            v16bf b;
            #pragma unroll
            for (int i = 0; i < 16; ++i) b[i] = (__bf16)wr[i];
            acc[j] = wmma_bf16(a, b, acc[j]);
        }
    }

    // Store: element (m = r + 8*half, n = lidx) of sub-tile j; head = tng.
    #pragma unroll
    for (int j = 0; j < 4; ++j) {
        const int ci = j * 16 + lidx;               // column within head
        const float bc = bias[tng * 64 + ci];
        #pragma unroll
        for (int r = 0; r < 8; ++r) {
            const int row = row0 + r + 8 * half;
            const int bb  = row >> 11, ss = row & (SEQ - 1);
            out[(((long)bb * N_HEADS + tng) * SEQ + ss) * D_K + ci] =
                (__bf16)(acc[j][r] + bc);
        }
    }
}

// ---------------------------------------------------------------------------
// Kernel 2: fused attention for one (b, h, 16-row tile).  4 waves.
// LDS: 16x2048 f32 scores + 128 f32 reduction + 2x 32x64 bf16 V staging.
// ---------------------------------------------------------------------------
__global__ __launch_bounds__(128)
void attn_fused_kernel(const __bf16* __restrict__ qh, const __bf16* __restrict__ kh,
                       const __bf16* __restrict__ vh,
                       float* __restrict__ attn,       // [b][h][s][s]
                       __bf16* __restrict__ ctx)       // [b][s][h*64+c]
{
    extern __shared__ float smem[];
    float*  sc     = smem;                   // 16 * 2048
    float*  red    = smem + 16 * SEQ;        // 128
    __bf16* vstage = (__bf16*)(red + 128);   // 2 * 2048 bf16 (8 KB)

    const int bid  = blockIdx.x;
    const int mt   = bid & 127;
    const int h    = (bid >> 7) & (N_HEADS - 1);
    const int b    = bid >> 11;
    const int lane = threadIdx.x & 31;
    const int wave = threadIdx.x >> 5;       // 0..3
    const int half = lane >> 4;
    const int lidx = lane & 15;

    const long base = ((long)b * N_HEADS + h) * SEQ * D_K;
    const int  row0 = mt * 16;

    // Q tile A-fragments (two K=32 chunks over d_k=64), pre-scaled 1/sqrt(64)
    v16bf aq0, aq1;
    {
        const __bf16* qrow = qh + base + (long)(row0 + lidx) * D_K;
        #pragma unroll
        for (int i = 0; i < 8; ++i) {
            aq0[i]     = (__bf16)((float)qrow[      16 * half + i] * 0.125f);
            aq0[8 + i] = (__bf16)((float)qrow[8 +  16 * half + i] * 0.125f);
            aq1[i]     = (__bf16)((float)qrow[32 + 16 * half + i] * 0.125f);
            aq1[8 + i] = (__bf16)((float)qrow[40 + 16 * half + i] * 0.125f);
        }
    }

    // ---- Phase 1: scores -> LDS ----------------------------------------
    for (int kt = wave; kt < SEQ / 16; kt += 4) {
        const __bf16* krow = kh + base + (long)(kt * 16 + lidx) * D_K;
        __builtin_prefetch(krow + 4 * 16 * D_K, 0, 3);       // next kt for wave
        const v16bf b0 = *(const v16bf*)(krow +      16 * half);
        const v16bf b1 = *(const v16bf*)(krow + 32 + 16 * half);
        v8f s = {};
        s = wmma_bf16(aq0, b0, s);
        s = wmma_bf16(aq1, b1, s);
        #pragma unroll
        for (int r = 0; r < 8; ++r)
            sc[(r + 8 * half) * SEQ + kt * 16 + lidx] = s[r];
    }
    __syncthreads();

    // ---- Phase 2: row softmax in LDS ------------------------------------
    const int srow = threadIdx.x >> 3;       // 16 rows, 8 threads/row
    const int sseg = threadIdx.x & 7;
    float* rowp = sc + srow * SEQ;

    float mx = -3.0e38f;
    for (int c = sseg * 256; c < sseg * 256 + 256; ++c) mx = fmaxf(mx, rowp[c]);
    red[threadIdx.x] = mx;
    __syncthreads();
    float m = red[srow * 8];
    #pragma unroll
    for (int i = 1; i < 8; ++i) m = fmaxf(m, red[srow * 8 + i]);

    float psum = 0.f;
    for (int c = sseg * 256; c < sseg * 256 + 256; ++c) {
        float e = __expf(rowp[c] - m);
        rowp[c] = e;
        psum += e;
    }
    __syncthreads();
    red[threadIdx.x] = psum;
    __syncthreads();
    float tot = red[srow * 8];
    #pragma unroll
    for (int i = 1; i < 8; ++i) tot += red[srow * 8 + i];
    const float inv = 1.0f / tot;
    for (int c = sseg * 256; c < sseg * 256 + 256; ++c) rowp[c] *= inv;
    __syncthreads();

    // Write-once attn tile: non-temporal streaming store (don't pollute L2)
    {
        const long abase = (((long)b * N_HEADS + h) * SEQ + row0) * SEQ;
        for (int idx = threadIdx.x; idx < 16 * SEQ; idx += 128)
            __builtin_nontemporal_store(sc[idx], &attn[abase + idx]);
    }

    // ---- Phase 3: out = P @ V ; wave owns 16 of 64 output columns -------
    const int dcol = wave * 16 + lidx;
    v8f o = {};
#if HAS_ASYNC_LDS
    const __bf16* vsrc = vh + base;          // keys contiguous, 128 B/row
    stage_v(vsrc, vstage, threadIdx.x);      // chunk 0 -> buffer 0
    for (int kc = 0; kc < SEQ / 32; ++kc) {
        wait_async0();
        __syncthreads();                     // current chunk resident in LDS
        if (kc + 1 < SEQ / 32)
            stage_v(vsrc + (long)(kc + 1) * 32 * D_K,
                    vstage + ((kc + 1) & 1) * 2048, threadIdx.x);
        const __bf16* vb = vstage + (kc & 1) * 2048;
        v16bf a, bv_;
        const float* prow = sc + lidx * SEQ + kc * 32;
        #pragma unroll
        for (int i = 0; i < 8; ++i) {
            a[i]     = (__bf16)prow[      16 * half + i];
            a[8 + i] = (__bf16)prow[8 +  16 * half + i];
        }
        #pragma unroll
        for (int i = 0; i < 16; ++i) bv_[i] = vb[(16 * half + i) * D_K + dcol];
        o = wmma_bf16(a, bv_, o);
        __syncthreads();                     // reads done before re-staging
    }
#else
    for (int kc = 0; kc < SEQ / 32; ++kc) {
        v16bf a, bv_;
        const float* prow = sc + lidx * SEQ + kc * 32;
        #pragma unroll
        for (int i = 0; i < 8; ++i) {
            a[i]     = (__bf16)prow[      16 * half + i];
            a[8 + i] = (__bf16)prow[8 +  16 * half + i];
        }
        const __bf16* vcol = vh + base + (long)(kc * 32 + 16 * half) * D_K + dcol;
        #pragma unroll
        for (int i = 0; i < 16; ++i) bv_[i] = vcol[(long)i * D_K];
        o = wmma_bf16(a, bv_, o);
    }
#endif
    #pragma unroll
    for (int r = 0; r < 8; ++r) {
        const int mrow = row0 + r + 8 * half;
        ctx[((long)b * SEQ + mrow) * D_MODEL + h * D_K + dcol] = (__bf16)o[r];
    }
}

// ---------------------------------------------------------------------------
// Kernel 3: output projection.  out = ctx(bf16) @ Wo^T + bo -> f32.
// One 16x64 tile per wave; 4096 wave-tiles.
// ---------------------------------------------------------------------------
__global__ __launch_bounds__(256)
void out_proj_kernel(const __bf16* __restrict__ ctx, const float* __restrict__ Wo,
                     const float* __restrict__ bo, float* __restrict__ out)
{
    const int lane = threadIdx.x & 31;
    const int wib  = threadIdx.x >> 5;
    const int t    = blockIdx.x * 8 + wib;          // 0..4095
    const int tm   = t >> 4;
    const int tng  = t & 15;
    const int row0 = tm * 16;
    const int col0 = tng * 64;
    const int half = lane >> 4;
    const int lidx = lane & 15;

    const __bf16* arow = ctx + (long)(row0 + lidx) * D_MODEL;
    const float*  wr0  = Wo + (long)(col0 + lidx) * D_MODEL + 16 * half;

    v8f acc[4] = {{}, {}, {}, {}};
    for (int kc = 0; kc < D_MODEL; kc += 32) {
        __builtin_prefetch(arow + kc + 64, 0, 3);
        const v8bf alo = *(const v8bf*)(arow + kc +     16 * half);
        const v8bf ahi = *(const v8bf*)(arow + kc + 8 + 16 * half);
        v16bf a;
        #pragma unroll
        for (int i = 0; i < 8; ++i) { a[i] = alo[i]; a[8 + i] = ahi[i]; }
        #pragma unroll
        for (int j = 0; j < 4; ++j) {
            const float* wr = wr0 + (long)j * 16 * D_MODEL + kc;
            __builtin_prefetch(wr + 32, 0, 3);
            v16bf b;
            #pragma unroll
            for (int i = 0; i < 16; ++i) b[i] = (__bf16)wr[i];
            acc[j] = wmma_bf16(a, b, acc[j]);
        }
    }
    #pragma unroll
    for (int j = 0; j < 4; ++j) {
        const int col = col0 + j * 16 + lidx;
        const float bc = bo[col];
        #pragma unroll
        for (int r = 0; r < 8; ++r) {
            const int row = row0 + r + 8 * half;
            out[(long)row * D_MODEL + col] = acc[j][r] + bc;
        }
    }
}

// ---------------------------------------------------------------------------
extern "C" void kernel_launch(void* const* d_in, const int* in_sizes, int n_in,
                              void* d_out, int out_size, void* d_ws, size_t ws_size,
                              hipStream_t stream)
{
    const float* q  = (const float*)d_in[0];
    const float* k  = (const float*)d_in[1];
    const float* v  = (const float*)d_in[2];
    const float* Wq = (const float*)d_in[3];
    const float* bq = (const float*)d_in[4];
    const float* Wk = (const float*)d_in[5];
    const float* bk = (const float*)d_in[6];
    const float* Wv = (const float*)d_in[7];
    const float* bv = (const float*)d_in[8];
    const float* Wo = (const float*)d_in[9];
    const float* bo = (const float*)d_in[10];

    float* out_proj = (float*)d_out;                               // 4,194,304 f32
    float* attn     = (float*)d_out + (long)BATCH * SEQ * D_MODEL; // 134,217,728 f32

    const long NELT = (long)BATCH * SEQ * D_MODEL;                 // 4,194,304
    __bf16* qh  = (__bf16*)d_ws;
    __bf16* kh  = qh + NELT;
    __bf16* vh  = kh + NELT;
    __bf16* ctx = vh + NELT;                                       // total 32 MB

    // 1) Q/K/V projections: 3*4096 wave-tiles, 8 waves per block
    proj_qkv_kernel<<<1536, 256, 0, stream>>>(q, k, v, Wq, bq, Wk, bk, Wv, bv,
                                              qh, kh, vh);

    // 2) fused attention: one WG per (b,h,16-row tile)
    const size_t lds = (size_t)(16 * SEQ + 128) * sizeof(float) + 2 * 2048 * 2;
    attn_fused_kernel<<<BATCH * N_HEADS * (SEQ / 16), 128, lds, stream>>>(
        qh, kh, vh, attn, ctx);

    // 3) output projection: 4096 wave-tiles
    out_proj_kernel<<<512, 256, 0, stream>>>(ctx, Wo, bo, out_proj);
}
